// CausalSelfAttention_75333726372170
// MI455X (gfx1250) — matrix operand order repaired and checked by reading, not compiled, not run on hardware
//
#include <hip/hip_runtime.h>

// CDNA5 (gfx1250) wave32 flash-attention with clipped softmax.
// bf16 WMMA 16x16x32, f32 accumulate. Two-pass (clip needs final softmax).

typedef __attribute__((ext_vector_type(16))) __bf16       v16bf;
typedef __attribute__((ext_vector_type(8)))  float        v8f;
typedef __attribute__((ext_vector_type(8)))  unsigned int v8u;

#define B_    2
#define S_    2048
#define H_    16
#define D_    128
#define QTILE 128
#define KTILE 32

// round-to-nearest-even f32 -> bf16 bits (inputs are finite)
__device__ __forceinline__ unsigned int f2bf(float f) {
  unsigned int u = __builtin_bit_cast(unsigned int, f);
  u += 0x7FFFu + ((u >> 16) & 1u);
  return u >> 16;
}

__device__ __forceinline__ v16bf pack8(const unsigned int* u) {
  v8u t;
  t[0] = u[0]; t[1] = u[1]; t[2] = u[2]; t[3] = u[3];
  t[4] = u[4]; t[5] = u[5]; t[6] = u[6]; t[7] = u[7];
  return __builtin_bit_cast(v16bf, t);
}

__global__ __launch_bounds__(256)
void fa_clip_kernel(const float* __restrict__ Q, const float* __restrict__ K,
                    const float* __restrict__ V, float* __restrict__ O) {
  __shared__ unsigned short sQ[QTILE * D_];      // 32 KB, Q tile bf16 (row major, pre-scaled)
  __shared__ unsigned short sK[KTILE * D_];      //  8 KB, K tile bf16 (row major, [kv][d])
  __shared__ unsigned short sV[D_ * KTILE];      //  8 KB, V tile bf16 transposed ([d][kv])
  __shared__ unsigned short sP[8 * 16 * KTILE];  //  8 KB, per-wave prob scratch

  const int tid  = threadIdx.x;
  const int w    = tid >> 5;        // wave id 0..7
  const int lane = tid & 31;
  const int hi   = lane >> 4;       // half-wave select
  const int ln   = lane & 15;

  const int b  = blockIdx.y / H_;
  const int h  = blockIdx.y % H_;
  const int q0 = blockIdx.x * QTILE;

  const float scale = 0.08838834764831845f;      // 1/sqrt(128), folded into Q

  // ---------------- stage Q tile -> LDS (bf16, pre-scaled), vectorized ----------------
  for (int i = tid; i < QTILE * D_ / 4; i += 256) {
    int r = i >> 5;                 // 32 float4 per row
    int c = (i & 31) * 4;
    const float4 q4 = *(const float4*)&Q[((size_t)(b * S_ + q0 + r) * H_ + h) * D_ + c];
    uint2 p;
    p.x = f2bf(q4.x * scale) | (f2bf(q4.y * scale) << 16);
    p.y = f2bf(q4.z * scale) | (f2bf(q4.w * scale) << 16);
    *(uint2*)&sQ[r * D_ + c] = p;
  }
  __syncthreads();

  // ---------------- gather Q A-fragments (16x32 per chunk, 4 chunks over D) ----
  v16bf qf[4];
  {
    const int row = w * 16 + ln;    // A layout: M = lane%16 in both halves
#pragma unroll
    for (int c = 0; c < 4; ++c) {
      unsigned int u[8];
#pragma unroll
      for (int v = 0; v < 8; ++v) {
        int kk = 2 * v + 16 * hi;   // contiguous K: k = 16*hi + s
        u[v] = *(const unsigned int*)&sQ[row * D_ + c * 32 + kk];
      }
      qf[c] = pack8(u);
    }
  }

  const int kv_end = q0 + QTILE;    // causal: no keys beyond the q tile
  const int qrow   = q0 + w * 16 + 8 * hi;  // + r gives this lane's global q row

  // prefetch address pattern: 32 rows x 8 x 64B chunks = one 32x128 f32 tile
  const int pf_r = tid >> 3;
  const int pf_c = (tid & 7) * 16;

  // =================== PASS 1: online softmax stats (m, l) ===================
  float m[8], l[8];
#pragma unroll
  for (int r = 0; r < 8; ++r) { m[r] = -1e30f; l[r] = 0.0f; }

  for (int kt = 0; kt < kv_end; kt += KTILE) {
    __syncthreads();
    for (int i = tid; i < KTILE * D_ / 4; i += 256) {
      int r = i >> 5;
      int c = (i & 31) * 4;
      const float4 k4 = *(const float4*)&K[((size_t)(b * S_ + kt + r) * H_ + h) * D_ + c];
      uint2 p;
      p.x = f2bf(k4.x) | (f2bf(k4.y) << 16);
      p.y = f2bf(k4.z) | (f2bf(k4.w) << 16);
      *(uint2*)&sK[r * D_ + c] = p;
    }
    __syncthreads();

    if (kt + KTILE < kv_end) {      // uniform branch: prefetch next K tile
      __builtin_prefetch(&K[((size_t)(b * S_ + kt + KTILE + pf_r) * H_ + h) * D_ + pf_c], 0, 1);
    }

    v8f sa[2];
#pragma unroll
    for (int nt = 0; nt < 2; ++nt) {
      v8f acc = {0.f, 0.f, 0.f, 0.f, 0.f, 0.f, 0.f, 0.f};
#pragma unroll
      for (int c = 0; c < 4; ++c) {
        unsigned int u[8];
#pragma unroll
        for (int j = 0; j < 8; ++j) {
          int kv = nt * 16 + ln;                 // B layout: N = lane%16
          int d  = c * 32 + 2 * j + 16 * hi;     // K pairs packed, halves split K
          u[j] = *(const unsigned int*)&sK[kv * D_ + d];
        }
        v16bf kf = pack8(u);
        acc = __builtin_amdgcn_wmma_f32_16x16x32_bf16(
                  false, qf[c], false, kf, (short)0, acc, false, false);
      }
      sa[nt] = acc;
    }

#pragma unroll
    for (int r = 0; r < 8; ++r) {
      int   qg = qrow + r;
      float s0 = sa[0][r];
      float s1 = sa[1][r];
      if (kt + ln      > qg) s0 = -1e30f;        // causal mask (select, no branch)
      if (kt + 16 + ln > qg) s1 = -1e30f;
      float mx = fmaxf(s0, s1);
#pragma unroll
      for (int off = 1; off < 16; off <<= 1) mx = fmaxf(mx, __shfl_xor(mx, off));
      float mn = fmaxf(m[r], mx);
      float p  = __expf(s0 - mn) + __expf(s1 - mn);
#pragma unroll
      for (int off = 1; off < 16; off <<= 1) p += __shfl_xor(p, off);
      l[r] = l[r] * __expf(m[r] - mn) + p;
      m[r] = mn;
    }
  }

  float invl[8];
#pragma unroll
  for (int r = 0; r < 8; ++r) invl[r] = 1.0f / l[r];  // diagonal always visible => l > 0

  // =================== PASS 2: clipped probs, P·V accumulate ==================
  v8f o[8];
#pragma unroll
  for (int t = 0; t < 8; ++t) o[t] = (v8f){0.f, 0.f, 0.f, 0.f, 0.f, 0.f, 0.f, 0.f};

  unsigned short* pw = &sP[w * 16 * KTILE];      // per-wave private prob scratch

  for (int kt = 0; kt < kv_end; kt += KTILE) {
    __syncthreads();
    for (int i = tid; i < KTILE * D_ / 4; i += 256) {
      int r = i >> 5;
      int c = (i & 31) * 4;
      size_t gi = ((size_t)(b * S_ + kt + r) * H_ + h) * D_ + c;
      const float4 k4 = *(const float4*)&K[gi];
      const float4 v4 = *(const float4*)&V[gi];
      uint2 p;
      p.x = f2bf(k4.x) | (f2bf(k4.y) << 16);
      p.y = f2bf(k4.z) | (f2bf(k4.w) << 16);
      *(uint2*)&sK[r * D_ + c] = p;
      sV[(c + 0) * KTILE + r] = (unsigned short)f2bf(v4.x);  // transpose for B-fragments
      sV[(c + 1) * KTILE + r] = (unsigned short)f2bf(v4.y);
      sV[(c + 2) * KTILE + r] = (unsigned short)f2bf(v4.z);
      sV[(c + 3) * KTILE + r] = (unsigned short)f2bf(v4.w);
    }
    __syncthreads();

    if (kt + KTILE < kv_end) {      // uniform branch: prefetch next K+V tiles
      size_t gp = ((size_t)(b * S_ + kt + KTILE + pf_r) * H_ + h) * D_ + pf_c;
      __builtin_prefetch(&K[gp], 0, 1);
      __builtin_prefetch(&V[gp], 0, 1);
    }

    // ---- recompute scores ----
    v8f sa[2];
#pragma unroll
    for (int nt = 0; nt < 2; ++nt) {
      v8f acc = {0.f, 0.f, 0.f, 0.f, 0.f, 0.f, 0.f, 0.f};
#pragma unroll
      for (int c = 0; c < 4; ++c) {
        unsigned int u[8];
#pragma unroll
        for (int j = 0; j < 8; ++j) {
          int kv = nt * 16 + ln;
          int d  = c * 32 + 2 * j + 16 * hi;
          u[j] = *(const unsigned int*)&sK[kv * D_ + d];
        }
        v16bf kf = pack8(u);
        acc = __builtin_amdgcn_wmma_f32_16x16x32_bf16(
                  false, qf[c], false, kf, (short)0, acc, false, false);
      }
      sa[nt] = acc;
    }

    // ---- normalized + clipped probs -> LDS (C/D layout -> row major bf16) ----
#pragma unroll
    for (int nt = 0; nt < 2; ++nt) {
#pragma unroll
      for (int r = 0; r < 8; ++r) {
        int   qg = qrow + r;
        int   kg = kt + nt * 16 + ln;
        float s  = sa[nt][r];
        if (kg > qg) s = -1e30f;
        float p  = __expf(s - m[r]) * invl[r];
        p = fminf(1.0f, fmaxf(0.0f, 1.06f * p - 0.03f));   // clip((z-g)p + g, 0, 1)
        pw[(r + 8 * hi) * KTILE + nt * 16 + ln] = (unsigned short)f2bf(p);
      }
    }
    asm volatile("s_wait_dscnt 0" ::: "memory");  // wave-local LDS store->load ordering

    // ---- P A-fragment (16x32) ----
    unsigned int up[8];
#pragma unroll
    for (int v = 0; v < 8; ++v) {
      int kk = 2 * v + 16 * hi;
      up[v] = *(const unsigned int*)&pw[ln * KTILE + kk];
    }
    v16bf pf = pack8(up);

    // ---- o[t] += P (16x32) x V (32x16) for each 16-wide D tile ----
#pragma unroll
    for (int t = 0; t < 8; ++t) {
      unsigned int u[8];
#pragma unroll
      for (int j = 0; j < 8; ++j) {
        int dcol = t * 16 + ln;
        int kv   = 2 * j + 16 * hi;
        u[j] = *(const unsigned int*)&sV[dcol * KTILE + kv];
      }
      v16bf vf = pack8(u);
      o[t] = __builtin_amdgcn_wmma_f32_16x16x32_bf16(
                 false, pf, false, vf, (short)0, o[t], false, false);
    }
  }

  // ---------------- writeback (f32, coalesced across lanes) ----------------
#pragma unroll
  for (int t = 0; t < 8; ++t) {
#pragma unroll
    for (int r = 0; r < 8; ++r) {
      int qg = qrow + r;
      int d  = t * 16 + ln;
      O[((size_t)(b * S_ + qg) * H_ + h) * D_ + d] = o[t][r];
    }
  }
}

extern "C" void kernel_launch(void* const* d_in, const int* in_sizes, int n_in,
                              void* d_out, int out_size, void* d_ws, size_t ws_size,
                              hipStream_t stream) {
  (void)in_sizes; (void)n_in; (void)out_size; (void)d_ws; (void)ws_size;
  const float* Q = (const float*)d_in[0];
  const float* K = (const float*)d_in[1];
  const float* V = (const float*)d_in[2];
  float* O = (float*)d_out;
  dim3 grid(S_ / QTILE, B_ * H_, 1);
  fa_clip_kernel<<<grid, dim3(256, 1, 1), 0, stream>>>(Q, K, V, O);
}
